// MultiHeadSelfAttention_91044716741059
// MI455X (gfx1250) — matrix operand order repaired and checked by reading, not compiled, not run on hardware
//
#include <hip/hip_runtime.h>
#include <hip/hip_bf16.h>

// ---------------------------------------------------------------------------
// Multi-head self-attention for MI455X (gfx1250), wave32 + WMMA f16->f32.
// Pipeline: [cvt weights f32->f16] -> [QKV projection GEMMs] ->
//           [flash attention (QK^T, online softmax, PV)] -> [out proj + bias]
// All matrix math runs on v_wmma_f32_16x16x32_f16.
// Attention streams K/V in 64-row blocks: softmax cross-lane reductions and
// accumulator rescales are amortized over 16 WMMAs per block.
// ---------------------------------------------------------------------------

#define NBATCH 2
#define SEQ    2048
#define EMBED  1024
#define HEADS  16
#define HDIM   64
#define QTILES (SEQ / 16)       // 128 q-row tiles per (n,h)

typedef __attribute__((ext_vector_type(16))) _Float16 v16h;
typedef __attribute__((ext_vector_type(8)))  _Float16 v8h;
typedef __attribute__((ext_vector_type(8)))  float    v8f;
typedef __attribute__((ext_vector_type(4)))  float    v4f;

#define DEVINL static __device__ __forceinline__

DEVINL v8f wmma_f16(v16h a, v16h b, v8f c) {
  // 8 args: (neg_a, A, neg_b, B, c_mod, C, reuse_a, reuse_b)
  return __builtin_amdgcn_wmma_f32_16x16x32_f16(false, a, false, b, (short)0, c,
                                                false, false);
}

// A fragment: 16x32 f16 tile, row-major source (stride in elements).
// Lane l: m = l&15, hi = l>>4. Halves 0..7 <- K = hi*8 + 0..7,
// halves 8..15 <- K = 16 + hi*8 + 0..7   (ISA 7.12.2, 16-bit A 16x32).
DEVINL v16h load_a_f16(const _Float16* base, int stride) {
  const int l  = threadIdx.x & 31;
  const int m  = l & 15, hi = l >> 4;
  const _Float16* p = base + (size_t)m * stride + hi * 8;
  v8h lo = *(const v8h*)(p);
  v8h hv = *(const v8h*)(p + 16);
  v16h r;
#pragma unroll
  for (int i = 0; i < 8; ++i) { r[i] = lo[i]; r[i + 8] = hv[i]; }
  return r;
}

// A fragment loaded from f32 source with convert-to-f16.
DEVINL v16h load_a_f32cvt(const float* base, int stride) {
  const int l  = threadIdx.x & 31;
  const int m  = l & 15, hi = l >> 4;
  const float* p = base + (size_t)m * stride + hi * 8;
  v4f f0 = *(const v4f*)(p);
  v4f f1 = *(const v4f*)(p + 4);
  v4f f2 = *(const v4f*)(p + 16);
  v4f f3 = *(const v4f*)(p + 20);
  v16h r;
#pragma unroll
  for (int i = 0; i < 4; ++i) {
    r[i]      = (_Float16)f0[i];
    r[i + 4]  = (_Float16)f1[i];
    r[i + 8]  = (_Float16)f2[i];
    r[i + 12] = (_Float16)f3[i];
  }
  return r;
}

// B fragment: 32x16 f16, element (k,n) taken from source stored [N][K]
// row-major (i.e. B^T layout). Lane l: n = l&15, halves 0..7 <- K = hi*8 + 0..7,
// halves 8..15 <- K = 16 + hi*8 + 0..7.
DEVINL v16h load_b_f16(const _Float16* base, int stride) {
  const int l  = threadIdx.x & 31;
  const int n  = l & 15, hi = l >> 4;
  const _Float16* p = base + (size_t)n * stride + hi * 8;
  v8h b0 = *(const v8h*)(p);
  v8h b1 = *(const v8h*)(p + 16);
  v16h r;
#pragma unroll
  for (int i = 0; i < 8; ++i) { r[i] = b0[i]; r[i + 8] = b1[i]; }
  return r;
}

// ---------------------------------------------------------------------------
// Kernel 0: weights f32 -> f16 (Wq/Wk/Wv 64x64, Wo 1024x1024).
// ---------------------------------------------------------------------------
__global__ __launch_bounds__(512) void mha_cvt_weights(
    const float* __restrict__ Wv, const float* __restrict__ Wk,
    const float* __restrict__ Wq, const float* __restrict__ Wo,
    _Float16* __restrict__ wv16, _Float16* __restrict__ wk16,
    _Float16* __restrict__ wq16, _Float16* __restrict__ wo16) {
  const int i = blockIdx.x * blockDim.x + threadIdx.x;
  if (i < HDIM * HDIM) {
    wv16[i] = (_Float16)Wv[i];
    wk16[i] = (_Float16)Wk[i];
    wq16[i] = (_Float16)Wq[i];
  }
  if (i < EMBED * EMBED) wo16[i] = (_Float16)Wo[i];
}

// ---------------------------------------------------------------------------
// Kernel 1: QKV head projections. One wave computes a 16(l) x 64(d) tile of
// one projected matrix. blockIdx.y selects {V, K, Q}. D = X_head(16x64) * W^T.
// W row-major [d][e] is directly the B-fragment layout (B(e,d) = W[d][e]).
// V is written transposed as vt[(n,h,d)][l] for the PV GEMM.
// ---------------------------------------------------------------------------
__global__ __launch_bounds__(128) void mha_proj_qkv(
    const float* __restrict__ values, const float* __restrict__ keys,
    const float* __restrict__ query,
    const _Float16* __restrict__ wv16, const _Float16* __restrict__ wk16,
    const _Float16* __restrict__ wq16,
    _Float16* __restrict__ vt16, _Float16* __restrict__ k16,
    _Float16* __restrict__ q16) {
  const int widb = threadIdx.x >> 5;
  const int wave = blockIdx.x * (blockDim.x >> 5) + widb;  // 0..4095
  const int qt = wave & (QTILES - 1);
  const int h  = (wave >> 7) & (HEADS - 1);
  const int n  = wave >> 11;
  const int which = blockIdx.y;  // 0=V 1=K 2=Q

  const float* X; const _Float16* W;
  if (which == 0)      { X = values; W = wv16; }
  else if (which == 1) { X = keys;   W = wk16; }
  else                 { X = query;  W = wq16; }

  const int l0 = qt * 16;
  const float* abase = X + ((size_t)(n * SEQ + l0)) * EMBED + h * HDIM;
  const v16h a0 = load_a_f32cvt(abase, EMBED);       // e in [0,32)
  const v16h a1 = load_a_f32cvt(abase + 32, EMBED);  // e in [32,64)

  const int lane = threadIdx.x & 31, nl = lane & 15, hi = lane >> 4;

#pragma unroll
  for (int dt = 0; dt < 4; ++dt) {
    const v16h b0 = load_b_f16(W + dt * 16 * HDIM, HDIM);
    const v16h b1 = load_b_f16(W + dt * 16 * HDIM + 32, HDIM);
    v8f c = {};
    c = wmma_f16(a0, b0, c);
    c = wmma_f16(a1, b1, c);
    if (which == 0) {
      // transposed: vt[((n*H+h)*64 + d) * SEQ + l], contiguous in l per lane
      _Float16* o = vt16 +
          ((size_t)((n * HEADS + h) * HDIM + dt * 16 + nl)) * SEQ + l0 + 8 * hi;
      v8h pk;
#pragma unroll
      for (int r = 0; r < 8; ++r) pk[r] = (_Float16)c[r];
      *(v8h*)o = pk;
    } else {
      _Float16* dst = (which == 1) ? k16 : q16;
      _Float16* o = dst + ((size_t)((n * HEADS + h) * SEQ + l0)) * HDIM +
                    dt * 16 + nl;
#pragma unroll
      for (int r = 0; r < 8; ++r) o[(size_t)(r + 8 * hi) * HDIM] = (_Float16)c[r];
    }
  }
}

// ---------------------------------------------------------------------------
// Kernel 2: flash attention. One wave owns a 16-row Q tile of one (n,h);
// streams K/V in 64-row blocks (4 k-tiles), online softmax once per block,
// accumulates O in f32. 16 WMMAs per block (8 for S, 8 for PV).
// ---------------------------------------------------------------------------
__global__ __launch_bounds__(128) void mha_attention(
    const _Float16* __restrict__ q16, const _Float16* __restrict__ k16,
    const _Float16* __restrict__ vt16, const int* __restrict__ mask,
    _Float16* __restrict__ o16) {
  __shared__ __align__(16) _Float16 pstage[4][16 * 64];  // 16x64 P per wave

  const int widb = threadIdx.x >> 5;
  const int wave = blockIdx.x * 4 + widb;  // 0..4095
  const int qt = wave & (QTILES - 1);
  const int h  = (wave >> 7) & (HEADS - 1);
  const int n  = wave >> 11;
  const int lane = threadIdx.x & 31, nl = lane & 15, hi = lane >> 4;
  const int l0 = qt * 16;

  const _Float16* qb = q16 + ((size_t)((n * HEADS + h) * SEQ + l0)) * HDIM;
  const v16h qa0 = load_a_f16(qb, HDIM);
  const v16h qa1 = load_a_f16(qb + 32, HDIM);

  const _Float16* kb = k16 + ((size_t)(n * HEADS + h)) * SEQ * HDIM;
  const _Float16* vb = vt16 + ((size_t)(n * HEADS + h)) * HDIM * SEQ;
  const int* mrow = mask + (size_t)n * SEQ;

  float mi[8], li[8], alpha[8];
  v8f acc[4];
#pragma unroll
  for (int r = 0; r < 8; ++r) { mi[r] = -__builtin_inff(); li[r] = 0.0f; }
#pragma unroll
  for (int dt = 0; dt < 4; ++dt) acc[dt] = (v8f){};

  const float scale = 0.03125f;  // 1/sqrt(EMBED) = 1/32

  for (int k0 = 0; k0 < SEQ; k0 += 64) {
    if (k0 + 64 < SEQ) {  // prefetch next K/V blocks (global_prefetch_b8)
      __builtin_prefetch(kb + (size_t)(k0 + 64 + nl) * HDIM, 0, 1);
      __builtin_prefetch(vb + (size_t)nl * SEQ + k0 + 64, 0, 1);
    }
    // ---- S = Q * K^T : four 16x16 tiles (k-dim split over d in two WMMAs)
    v8f s[4];
#pragma unroll
    for (int jt = 0; jt < 4; ++jt) {
      const _Float16* kt = kb + (size_t)(k0 + jt * 16) * HDIM;
      const v16h b0 = load_b_f16(kt, HDIM);
      const v16h b1 = load_b_f16(kt + 32, HDIM);
      v8f c = {};
      c = wmma_f16(qa0, b0, c);
      c = wmma_f16(qa1, b1, c);
      const int mv = mrow[k0 + jt * 16 + nl];
#pragma unroll
      for (int r = 0; r < 8; ++r) s[jt][r] = mv ? c[r] * scale : -1e20f;
    }
    // ---- online softmax stats, once per 64 columns; rows live in 16-lane
    //      groups (C layout: vgpr r <-> row r+8*hi, lanes <-> columns)
#pragma unroll
    for (int r = 0; r < 8; ++r) {
      float mx = fmaxf(fmaxf(s[0][r], s[1][r]), fmaxf(s[2][r], s[3][r]));
      mx = fmaxf(mx, __shfl_xor(mx, 1, 16));
      mx = fmaxf(mx, __shfl_xor(mx, 2, 16));
      mx = fmaxf(mx, __shfl_xor(mx, 4, 16));
      mx = fmaxf(mx, __shfl_xor(mx, 8, 16));
      const float mnew = fmaxf(mi[r], mx);
      const float a = __expf(mi[r] - mnew);
      mi[r] = mnew;
      float ps = 0.0f;
#pragma unroll
      for (int jt = 0; jt < 4; ++jt) {
        const float e = __expf(s[jt][r] - mnew);
        s[jt][r] = e;
        ps += e;
      }
      ps += __shfl_xor(ps, 1, 16);
      ps += __shfl_xor(ps, 2, 16);
      ps += __shfl_xor(ps, 4, 16);
      ps += __shfl_xor(ps, 8, 16);
      li[r] = li[r] * a + ps;
      alpha[r] = a;
    }
#pragma unroll
    for (int dt = 0; dt < 4; ++dt)
#pragma unroll
      for (int r = 0; r < 8; ++r) acc[dt][r] *= alpha[r];

    // ---- C-layout -> A-layout for P via per-wave LDS tile (same-wave DS
    //      ops are processed in order; compiler inserts s_wait_dscnt).
    _Float16* ps_ = pstage[widb];
#pragma unroll
    for (int jt = 0; jt < 4; ++jt)
#pragma unroll
      for (int r = 0; r < 8; ++r)
        ps_[(r + 8 * hi) * 64 + jt * 16 + nl] = (_Float16)s[jt][r];
    const v16h pa0 = load_a_f16(ps_, 64);       // k in [0,32)
    const v16h pa1 = load_a_f16(ps_ + 32, 64);  // k in [32,64)

    // ---- O += P * V : V^T layout [d][l] is directly the B-fragment source
#pragma unroll
    for (int dt = 0; dt < 4; ++dt) {
      const _Float16* vd = vb + (size_t)(dt * 16) * SEQ + k0;
      acc[dt] = wmma_f16(pa0, load_b_f16(vd, SEQ), acc[dt]);
      acc[dt] = wmma_f16(pa1, load_b_f16(vd + 32, SEQ), acc[dt]);
    }
  }

  // ---- normalize and emit f16 rows [n*SEQ+l][EMBED] for the output GEMM
  float invl[8];
#pragma unroll
  for (int r = 0; r < 8; ++r) invl[r] = 1.0f / li[r];
  _Float16* ob = o16 + ((size_t)(n * SEQ + l0)) * EMBED + h * HDIM;
#pragma unroll
  for (int dt = 0; dt < 4; ++dt)
#pragma unroll
    for (int r = 0; r < 8; ++r)
      ob[(size_t)(r + 8 * hi) * EMBED + dt * 16 + nl] =
          (_Float16)(acc[dt][r] * invl[r]);
}

// ---------------------------------------------------------------------------
// Kernel 3: output projection Y = O * Wo^T + bo (4096 x 1024 x 1024).
// One wave computes a 16(row) x 64(col) tile; K loop over 1024 in 32 steps.
// ---------------------------------------------------------------------------
__global__ __launch_bounds__(128) void mha_out_proj(
    const _Float16* __restrict__ o16, const _Float16* __restrict__ wo16,
    const float* __restrict__ bo, float* __restrict__ out) {
  const int wave = blockIdx.x * 4 + (threadIdx.x >> 5);  // 0..4095
  const int ct = wave & 15;        // 16 column tiles of 64
  const int rt = wave >> 4;        // 256 row tiles of 16
  const int lane = threadIdx.x & 31, nl = lane & 15, hi = lane >> 4;

  v8f acc[4];
#pragma unroll
  for (int dt = 0; dt < 4; ++dt) acc[dt] = (v8f){};

  const _Float16* ab = o16 + (size_t)rt * 16 * EMBED;
  for (int e0 = 0; e0 < EMBED; e0 += 32) {
    const v16h a = load_a_f16(ab + e0, EMBED);
#pragma unroll
    for (int dt = 0; dt < 4; ++dt) {
      const v16h b =
          load_b_f16(wo16 + (size_t)(ct * 64 + dt * 16) * EMBED + e0, EMBED);
      acc[dt] = wmma_f16(a, b, acc[dt]);
    }
  }
#pragma unroll
  for (int dt = 0; dt < 4; ++dt) {
    const int col = ct * 64 + dt * 16 + nl;
    const float bb = bo[col];
    float* o = out + (size_t)(rt * 16) * EMBED + col;
#pragma unroll
    for (int r = 0; r < 8; ++r)
      o[(size_t)(r + 8 * hi) * EMBED] = acc[dt][r] + bb;
  }
}

// ---------------------------------------------------------------------------
// Host launcher. Workspace layout (bytes), ~35.7 MB total:
//   [0)        wv16 8K | wk16 8K | wq16 8K | wo16 2M |
//   q16 8M | k16 8M | vt16 8M | o16(attn rows) 8M
// ---------------------------------------------------------------------------
extern "C" void kernel_launch(void* const* d_in, const int* in_sizes, int n_in,
                              void* d_out, int out_size, void* d_ws,
                              size_t ws_size, hipStream_t stream) {
  const float* values = (const float*)d_in[0];
  const float* keys   = (const float*)d_in[1];
  const float* query  = (const float*)d_in[2];
  const int*   mask   = (const int*)d_in[3];
  const float* Wv     = (const float*)d_in[4];
  const float* Wk     = (const float*)d_in[5];
  const float* Wq     = (const float*)d_in[6];
  const float* Wo     = (const float*)d_in[7];
  const float* bo     = (const float*)d_in[8];
  float* out = (float*)d_out;

  char* ws = (char*)d_ws;
  const size_t SMALLW = (size_t)HDIM * HDIM * sizeof(_Float16);   // 8192
  _Float16* wv16 = (_Float16*)(ws);
  _Float16* wk16 = (_Float16*)(ws + SMALLW);
  _Float16* wq16 = (_Float16*)(ws + 2 * SMALLW);
  _Float16* wo16 = (_Float16*)(ws + 3 * SMALLW);
  const size_t HEADSZ = (size_t)NBATCH * HEADS * SEQ * HDIM * sizeof(_Float16);
  size_t off = 3 * SMALLW + (size_t)EMBED * EMBED * sizeof(_Float16);
  _Float16* q16  = (_Float16*)(ws + off); off += HEADSZ;
  _Float16* k16  = (_Float16*)(ws + off); off += HEADSZ;
  _Float16* vt16 = (_Float16*)(ws + off); off += HEADSZ;
  _Float16* ao16 = (_Float16*)(ws + off);

  // 0) weight conversion (1M threads cover Wo exactly)
  mha_cvt_weights<<<2048, 512, 0, stream>>>(Wv, Wk, Wq, Wo, wv16, wk16, wq16,
                                            wo16);
  // 1) QKV projection: 4096 waves per matrix, y picks {V,K,Q}
  mha_proj_qkv<<<dim3(1024, 3, 1), 128, 0, stream>>>(
      values, keys, query, wv16, wk16, wq16, vt16, k16, q16);
  // 2) attention: 4096 waves (one per (n,h,16-row q tile)), 64-row k blocks
  mha_attention<<<1024, 128, 0, stream>>>(q16, k16, vt16, mask, ao16);
  // 3) output projection + bias: 4096 waves
  mha_out_proj<<<1024, 128, 0, stream>>>(ao16, wo16, bo, out);
}
